// ConstituencyTreeLSTM_90417651515480
// MI455X (gfx1250) — matrix operand order repaired and checked by reading, not compiled, not run on hardware
//
#include <hip/hip_runtime.h>
#include <hip/hip_bf16.h>

typedef __attribute__((ext_vector_type(16))) __bf16 v16bf;
typedef __attribute__((ext_vector_type(8)))  float  v8f;

#define B_SZ    256
#define N_NODES 256
#define K_DIM   512
#define D_MEM   512
#define NCOLS   2048  // 1536 iou cols + 512 f cols
#define CHUNK   32
#define MAXM    (CHUNK * B_SZ)  // 8192 rows max per chunk

__device__ __forceinline__ unsigned short f2bf(float f) {
  unsigned u = __float_as_uint(f);
  u += 0x7FFFu + ((u >> 16) & 1u);  // round to nearest even
  return (unsigned short)(u >> 16);
}

// ---------------------------------------------------------------------------
// WMMA fragment layouts (wave32, 16x16x32 bf16):
//  A frag (16 rows x 32 k): lane = (m%16) + 16*hi, hi=(k%16)/8,
//                           elem = (k/16)*8 + (k%8)
//  B frag (32 k x 16 n):    lane l, elem e -> n = l%16, k = (e/8)*16 + (l/16)*8 + (e%8)
// All packed buffers store one frag as 32 lanes x 16 contiguous bf16 (1 KB).
// ---------------------------------------------------------------------------

// Pack W = concat(Wiou[1536x512], Wf[512x512]) into B-fragment layout:
// frag (nt, kt) at ((nt*16 + kt) * 512).
__global__ __launch_bounds__(256)
void pack_weights(const float* __restrict__ Wiou, const float* __restrict__ Wf,
                  unsigned short* __restrict__ out) {
  int idx = blockIdx.x * 256 + threadIdx.x;  // < 2048*512
  int e      = idx & 15;
  int lane   = (idx >> 4) & 31;
  int fragid = idx >> 9;                     // nt*16 + kt (KT = 512/32 = 16)
  int kt = fragid & 15;
  int nt = fragid >> 4;
  int n = nt * 16 + (lane & 15);
  int k = kt * 32 + ((e >> 3) << 4) + ((lane >> 4) << 3) + (e & 7);
  float v = (n < 1536) ? Wiou[n * 512 + k] : Wf[(n - 1536) * 512 + k];
  out[idx] = f2bf(v);
}

// Pack inputs (B, N, 512) f32 into A-fragment bf16 layout keyed by
// (node, bgroup g, kt): frag at (((node*16 + g)*16 + kt) * 512).
// Rows of frag (node, g) are b = g*16 + (lane&15).
__global__ __launch_bounds__(256)
void pack_inputs(const float* __restrict__ inputs, unsigned short* __restrict__ out) {
  unsigned idx = blockIdx.x * 256 + threadIdx.x;  // < 256*16*16*512 = 33,554,432
  int e    = idx & 15;
  int lane = (idx >> 4) & 31;
  int kt   = (idx >> 9) & 15;
  int g    = (idx >> 13) & 15;
  int node = idx >> 17;
  int mloc = lane & 15, hi = lane >> 4;
  int k = kt * 32 + ((e >> 3) << 4) + (hi << 3) + (e & 7);
  int b = g * 16 + mloc;
  out[idx] = f2bf(inputs[(size_t)(b * N_NODES + node) * K_DIM + k]);
}

// ---------------------------------------------------------------------------
// Fragment-direct GEMM: Y[m, :2048] = A[m, :512] @ Wpack.T (bf16 WMMA, f32 acc)
// A comes pre-packed in fragment layout (Ipack for X side, Hpack for L/R).
// Row m -> node = node_base + m/256, b = m%256; L/R gather via child_idx.
// Inner loop: 12 global b128 loads + 8 WMMAs. No LDS, no barriers.
// ---------------------------------------------------------------------------
__global__ __launch_bounds__(256)
void gemm_frag(const unsigned short* __restrict__ Apack,
               const int* __restrict__ child_idx,
               int node_base,
               const unsigned short* __restrict__ Bpack,
               float* __restrict__ Yiou,
               float* __restrict__ Yf,
               int accum) {
  const int tid  = threadIdx.x;
  const int lane = tid & 31;
  const int wid  = tid >> 5;
  const int wm   = wid & 1;   // 2 M-strips of 64 rows
  const int wn   = wid >> 1;  // 4 N-strips of 32 cols

  const int mBlock = blockIdx.y * 128;
  const int nBlock = blockIdx.x * 128;

  const unsigned short* aptr[4];
  #pragma unroll
  for (int r = 0; r < 4; ++r) {
    int row  = mBlock + wm * 64 + r * 16;          // frag-aligned (all 16 rows same node)
    int node = node_base + (row >> 8);
    int g    = (row & 255) >> 4;
    int src  = child_idx ? child_idx[node] : node;
    aptr[r] = Apack + ((size_t)src * 16 + g) * 16 * 512 + lane * 16;
  }
  const unsigned short* bptr[2];
  #pragma unroll
  for (int cc = 0; cc < 2; ++cc) {
    int nt = (nBlock >> 4) + wn * 2 + cc;
    bptr[cc] = Bpack + (size_t)nt * 16 * 512 + lane * 16;
  }

  v8f acc[4][2];
  #pragma unroll
  for (int r = 0; r < 4; ++r)
    #pragma unroll
    for (int cc = 0; cc < 2; ++cc) {
      v8f z = {0.f,0.f,0.f,0.f,0.f,0.f,0.f,0.f};
      acc[r][cc] = z;
    }

  for (int kt = 0; kt < K_DIM / 32; ++kt) {
    v16bf bfr[2];
    #pragma unroll
    for (int cc = 0; cc < 2; ++cc) {
      bfr[cc] = *reinterpret_cast<const v16bf*>(bptr[cc] + kt * 512);
      if (kt + 1 < K_DIM / 32)
        __builtin_prefetch(bptr[cc] + (kt + 1) * 512, 0, 1);  // global_prefetch
    }
    #pragma unroll
    for (int r = 0; r < 4; ++r) {
      v16bf afr = *reinterpret_cast<const v16bf*>(aptr[r] + kt * 512);
      #pragma unroll
      for (int cc = 0; cc < 2; ++cc)
        acc[r][cc] = __builtin_amdgcn_wmma_f32_16x16x32_bf16(
            false, afr, false, bfr[cc], (short)0, acc[r][cc], false, false);
    }
  }

  // Epilogue. C/D layout: VGPR j, lanes 0-15 -> M=j, N=lane; lanes 16-31 -> M=j+8.
  const bool isIou = (nBlock < 1536);
  #pragma unroll
  for (int r = 0; r < 4; ++r) {
    int rowb = mBlock + wm * 64 + r * 16 + ((lane >> 4) << 3);
    #pragma unroll
    for (int cc = 0; cc < 2; ++cc) {
      int col = nBlock + wn * 32 + cc * 16 + (lane & 15);
      #pragma unroll
      for (int j = 0; j < 8; ++j) {
        int m = rowb + j;
        float v = acc[r][cc][j];
        if (isIou) {
          float* p = Yiou + (size_t)m * 1536 + col;
          *p = accum ? (*p + v) : v;
        } else {
          Yf[(size_t)m * 512 + (col - 1536)] = v;
        }
      }
    }
  }
}

// ---------------------------------------------------------------------------
// Gate epilogue: computes c,h; writes C (f32, for child cL/cR gathers),
// Hpack (fragment-swizzled bf16, consumed by parent-level GEMMs), and
// Hout (f32, node 0 only -> final output).
// ---------------------------------------------------------------------------
__global__ __launch_bounds__(256)
void node_pointwise(const float* __restrict__ Yiou, const float* __restrict__ YfX,
                    const float* __restrict__ YfL, const float* __restrict__ YfR,
                    float* __restrict__ C, unsigned short* __restrict__ Hpack,
                    float* __restrict__ Hout,
                    const float* __restrict__ b_ioux, const float* __restrict__ b_iouh,
                    const float* __restrict__ b_iouhL, const float* __restrict__ b_iouhR,
                    const float* __restrict__ b_fx, const float* __restrict__ b_fhL,
                    const float* __restrict__ b_fhR,
                    const int* __restrict__ left_idx, const int* __restrict__ right_idx,
                    const int* __restrict__ num_children,
                    int node_base, int total) {
  int idx = blockIdx.x * 256 + threadIdx.x;
  if (idx >= total) return;
  int d = idx & 511;
  int m = idx >> 9;
  int node = node_base + (m >> 8);
  int b = m & 255;
  int nc = num_children[node];

  size_t yb = (size_t)m * 1536;
  float gi = Yiou[yb + d]        + b_ioux[d];
  float go = Yiou[yb + 512 + d]  + b_ioux[512 + d];
  float gu = Yiou[yb + 1024 + d] + b_ioux[1024 + d];
  if (nc >= 1) { gi += b_iouhL[d]; go += b_iouhL[512 + d]; gu += b_iouhL[1024 + d]; }
  else         { gi += b_iouh[d];  go += b_iouh[512 + d];  gu += b_iouh[1024 + d]; }
  if (nc >= 2) { gi += b_iouhR[d]; go += b_iouhR[512 + d]; gu += b_iouhR[1024 + d]; }

  float ig = 1.f / (1.f + __expf(-gi));
  float og = 1.f / (1.f + __expf(-go));
  float ug = tanhf(gu);
  float c  = ig * ug;

  float fx = YfX[(size_t)m * 512 + d] + b_fx[d];
  if (nc >= 1) {
    int li = left_idx[node];
    float cL = C[(size_t)(b * 256 + li) * 512 + d];
    float fL = 1.f / (1.f + __expf(-(YfL[(size_t)m * 512 + d] + b_fhL[d] + fx)));
    c += fL * cL;
  }
  if (nc >= 2) {
    int ri = right_idx[node];
    float cR = C[(size_t)(b * 256 + ri) * 512 + d];
    float fR = 1.f / (1.f + __expf(-(YfR[(size_t)m * 512 + d] + b_fhR[d] + fx)));
    c += fR * cR;
  }
  float h = og * tanhf(c);

  C[(size_t)(b * 256 + node) * 512 + d] = c;

  // Fragment-swizzled bf16 h for the parent level's GEMM
  int g    = b >> 4;
  int mloc = b & 15;
  int kl   = d & 31;                       // k within frag
  int hi   = (kl & 15) >> 3;
  int e2   = ((kl >> 4) << 3) + (kl & 7);
  Hpack[(((size_t)node * 16 + g) * 16 + (d >> 5)) * 512 + (mloc + (hi << 4)) * 16 + e2]
      = f2bf(h);

  if (node == 0) Hout[(size_t)b * 512 + d] = h;
}

__global__ __launch_bounds__(256)
void copy_out(const float* __restrict__ C, const float* __restrict__ Hout,
              float* __restrict__ out) {
  int idx = blockIdx.x * 256 + threadIdx.x;  // < B*D = 131072
  int b = idx >> 9;
  int d = idx & 511;
  out[idx] = C[(size_t)(b * 256) * 512 + d];  // node 0
  out[131072 + idx] = Hout[idx];
}

// ---------------------------------------------------------------------------
extern "C" void kernel_launch(void* const* d_in, const int* in_sizes, int n_in,
                              void* d_out, int out_size, void* d_ws, size_t ws_size,
                              hipStream_t stream) {
  (void)in_sizes; (void)n_in; (void)out_size; (void)ws_size;
  const float* inputs  = (const float*)d_in[0];
  const float* W_ioux  = (const float*)d_in[1];
  const float* b_ioux  = (const float*)d_in[2];
  const float* b_iouh  = (const float*)d_in[4];   // W_iouh (d_in[3]) unused by ref
  const float* W_iouhL = (const float*)d_in[5];
  const float* b_iouhL = (const float*)d_in[6];
  const float* W_iouhR = (const float*)d_in[7];
  const float* b_iouhR = (const float*)d_in[8];
  const float* W_fx    = (const float*)d_in[9];
  const float* b_fx    = (const float*)d_in[10];
  const float* W_fhL   = (const float*)d_in[13];  // W_fh/b_fh (11,12) unused by ref
  const float* b_fhL   = (const float*)d_in[14];
  const float* W_fhR   = (const float*)d_in[15];
  const float* b_fhR   = (const float*)d_in[16];
  const int* left_idx  = (const int*)d_in[17];
  const int* right_idx = (const int*)d_in[18];
  const int* num_children = (const int*)d_in[19];

  char* ws = (char*)d_ws;
  size_t off = 0;
  auto alloc = [&](size_t bytes) -> void* {
    void* p = ws + off;
    off = (off + bytes + 255) & ~(size_t)255;
    return p;
  };
  const size_t PACK_ELEMS = (size_t)N_NODES * 16 * 16 * 512;         // 33.5M frag elems
  float* C    = (float*)alloc((size_t)B_SZ * N_NODES * D_MEM * 4);   // 128 MB
  float* Yiou = (float*)alloc((size_t)MAXM * 1536 * 4);              // 48 MB
  float* YfX  = (float*)alloc((size_t)MAXM * 512 * 4);
  float* YfL  = (float*)alloc((size_t)MAXM * 512 * 4);
  float* YfR  = (float*)alloc((size_t)MAXM * 512 * 4);
  float* Hout = (float*)alloc((size_t)B_SZ * D_MEM * 4);             // 512 KB
  unsigned short* Ipack = (unsigned short*)alloc(PACK_ELEMS * 2);    // 64 MB
  unsigned short* Hpack = (unsigned short*)alloc(PACK_ELEMS * 2);    // 64 MB
  unsigned short* WX = (unsigned short*)alloc((size_t)2048 * 512 * 2);
  unsigned short* WL = (unsigned short*)alloc((size_t)2048 * 512 * 2);
  unsigned short* WR = (unsigned short*)alloc((size_t)2048 * 512 * 2);

  hipMemsetAsync(C, 0, (size_t)B_SZ * N_NODES * D_MEM * 4, stream);
  hipMemsetAsync(Hpack, 0, PACK_ELEMS * 2, stream);   // absent child (idx 0) reads zeros
  pack_weights<<<4096, 256, 0, stream>>>(W_ioux, W_fx, WX);
  pack_weights<<<4096, 256, 0, stream>>>(W_iouhL, W_fhL, WL);
  pack_weights<<<4096, 256, 0, stream>>>(W_iouhR, W_fhR, WR);
  pack_inputs<<<(unsigned)(PACK_ELEMS / 256), 256, 0, stream>>>(inputs, Ipack);

  // Bottom-up over tree levels; same-level nodes are independent -> batched GEMMs.
  for (int lvl = 8; lvl >= 0; --lvl) {
    int start = (1 << lvl) - 1;
    int cnt = (1 << lvl);
    if (start + cnt > N_NODES) cnt = N_NODES - start;
    for (int cb = start; cb < start + cnt; cb += CHUNK) {
      int rem = start + cnt - cb;
      int P = (rem < CHUNK) ? rem : CHUNK;
      int M = P * B_SZ;                       // multiple of 256
      dim3 grid(NCOLS / 128, M / 128);
      gemm_frag<<<grid, 256, 0, stream>>>(Ipack, nullptr, cb, WX, Yiou, YfX, 0);
      bool anyL = (cb <= 127);                // node t has left child iff t <= 127
      bool anyR = (cb <= 126);
      if (anyL) gemm_frag<<<grid, 256, 0, stream>>>(Hpack, left_idx,  cb, WL, Yiou, YfL, 1);
      if (anyR) gemm_frag<<<grid, 256, 0, stream>>>(Hpack, right_idx, cb, WR, Yiou, YfR, 1);
      int total = M * D_MEM;
      node_pointwise<<<(total + 255) / 256, 256, 0, stream>>>(
          Yiou, YfX, YfL, YfR, C, Hpack, Hout,
          b_ioux, b_iouh, b_iouhL, b_iouhR, b_fx, b_fhL, b_fhR,
          left_idx, right_idx, num_children, cb, total);
    }
  }
  copy_out<<<(B_SZ * D_MEM + 255) / 256, 256, 0, stream>>>(C, Hout, (float*)d_out);
}
